// LSTM_77635828842725
// MI455X (gfx1250) — matrix operand order repaired and checked by reading, not compiled, not run on hardware
//
#include <hip/hip_runtime.h>
#include <hip/hip_bf16.h>

typedef __attribute__((ext_vector_type(16))) _Float16 v16h;
typedef __attribute__((ext_vector_type(8)))  float    v8f;

#define LSTM_B 32768
#define LSTM_T 28
#define LSTM_I 28
#define LSTM_H 64
#define LSTM_G 256   // 4*H
#define LSTM_O 10
#define BM     16    // batch tile per block (WMMA M)

__device__ __forceinline__ float fast_sigmoid(float x) {
    return __builtin_amdgcn_rcpf(1.0f + __expf(-x));
}
__device__ __forceinline__ float fast_tanh(float x) {
    // tanh(x) = 1 - 2/(exp(2x)+1)
    return 1.0f - 2.0f * __builtin_amdgcn_rcpf(__expf(2.0f * x) + 1.0f);
}

__global__ __launch_bounds__(512) void lstm_wmma_kernel(
    const float* __restrict__ x,      // [B, T, I]
    const float* __restrict__ W_ih,   // [4H, I]
    const float* __restrict__ W_hh,   // [4H, H]
    const float* __restrict__ b_ih,   // [4H]
    const float* __restrict__ b_hh,   // [4H]
    const float* __restrict__ W_out,  // [O, H]
    const float* __restrict__ b_out,  // [O]
    float* __restrict__ out)          // [B, O]
{
    __shared__ _Float16 h16[BM][LSTM_H];      // hidden state, f16 (WMMA A source)
    __shared__ float    c_st[BM][LSTM_H];     // cell state, f32
    __shared__ float    z_st[BM][LSTM_G];     // gate pre-activations, f32
    __shared__ _Float16 x16[BM][32];          // current x rows, padded K=28->32

    const int tid  = threadIdx.x;
    const int lane = tid & 31;
    const int w    = tid >> 5;         // wave 0..15 -> gate-column tile
    const int kh   = lane >> 4;        // lane half (K-split for A/B layouts)
    const int mrow = lane & 15;        // M row for A / N col within tile for B
    const int ncol = w * 16 + mrow;    // this lane's gate column (0..255)
    const int b0   = blockIdx.x * BM;  // batch base for this block

    // ---- Load per-wave resident B-tiles (contiguous K order, as in A: lane=N, VGPR j holds
    // K = kh*16 + 2j, 2j+1 within a 32-wide K window) -------------------------
    v16h Bx  = {};   // W_ih^T, K window [0,32) (28 real, 4 zero-padded)
    v16h Bh0 = {};   // W_hh^T, K window [0,32)
    v16h Bh1 = {};   // W_hh^T, K window [32,64)
    #pragma unroll
    for (int j = 0; j < 8; ++j) {
        const int k = kh * 16 + 2 * j;
        Bh0[2*j]   = (_Float16)W_hh[ncol * LSTM_H + k];
        Bh0[2*j+1] = (_Float16)W_hh[ncol * LSTM_H + k + 1];
        Bh1[2*j]   = (_Float16)W_hh[ncol * LSTM_H + 32 + k];
        Bh1[2*j+1] = (_Float16)W_hh[ncol * LSTM_H + 32 + k + 1];
        Bx[2*j]    = (k     < LSTM_I) ? (_Float16)W_ih[ncol * LSTM_I + k]     : (_Float16)0.0f;
        Bx[2*j+1]  = (k + 1 < LSTM_I) ? (_Float16)W_ih[ncol * LSTM_I + k + 1] : (_Float16)0.0f;
    }
    const float bias = b_ih[ncol] + b_hh[ncol];

    // ---- Zero-init h, c ------------------------------------------------------
    for (int i = tid; i < BM * LSTM_H; i += 512) {
        h16[i >> 6][i & 63] = (_Float16)0.0f;
        c_st[i >> 6][i & 63] = 0.0f;
    }
    __syncthreads();

    // ---- Time loop -----------------------------------------------------------
    for (int t = 0; t < LSTM_T; ++t) {
        // Stage x[:, t, :] into LDS as padded f16 (512 threads -> 16x32 cells)
        {
            const int m = tid >> 5;
            const int i = tid & 31;
            float xv = 0.0f;
            if (i < LSTM_I)
                xv = x[(size_t)(b0 + m) * (LSTM_T * LSTM_I) + t * LSTM_I + i];
            x16[m][i] = (_Float16)xv;
            if (t + 1 < LSTM_T && i < LSTM_I)
                __builtin_prefetch(x + (size_t)(b0 + m) * (LSTM_T * LSTM_I)
                                     + (t + 1) * LSTM_I + i, 0, 0);
        }
        __syncthreads();  // x16 ready; also orders prev-step h16 writes vs reads

        // Build A-tiles (contiguous K order, as in B: lane half selects K = kh*16 + s)
        v16h Ax, Ah0, Ah1;
        #pragma unroll
        for (int j = 0; j < 8; ++j) {
            const int kOff = kh * 16 + 2 * j;
            Ax[2*j]    = x16[mrow][kOff];
            Ax[2*j+1]  = x16[mrow][kOff + 1];
            Ah0[2*j]   = h16[mrow][kOff];
            Ah0[2*j+1] = h16[mrow][kOff + 1];
            Ah1[2*j]   = h16[mrow][32 + kOff];
            Ah1[2*j+1] = h16[mrow][32 + kOff + 1];
        }

        // z = bias + x W_ih^T + h W_hh^T  (f32 accumulate)
        v8f acc = {bias, bias, bias, bias, bias, bias, bias, bias};
        acc = __builtin_amdgcn_wmma_f32_16x16x32_f16(false, Ax,  false, Bx,  (short)0, acc, false, false);
        acc = __builtin_amdgcn_wmma_f32_16x16x32_f16(false, Ah0, false, Bh0, (short)0, acc, false, false);
        acc = __builtin_amdgcn_wmma_f32_16x16x32_f16(false, Ah1, false, Bh1, (short)0, acc, false, false);

        // Scatter C tile: VGPR r -> M = kh*8 + r, N = ncol
        #pragma unroll
        for (int r = 0; r < 8; ++r)
            z_st[kh * 8 + r][ncol] = acc[r];
        __syncthreads();

        // Elementwise gate math: 16x64 cells, 2 per thread (gate order i,f,g,o)
        for (int c = tid; c < BM * LSTM_H; c += 512) {
            const int m = c >> 6;
            const int k = c & 63;
            const float iv = z_st[m][k];
            const float fv = z_st[m][LSTM_H + k];
            const float gv = z_st[m][2 * LSTM_H + k];
            const float ov = z_st[m][3 * LSTM_H + k];
            const float cn = fast_sigmoid(fv) * c_st[m][k]
                           + fast_sigmoid(iv) * fast_tanh(gv);
            c_st[m][k] = cn;
            h16[m][k]  = (_Float16)(fast_sigmoid(ov) * fast_tanh(cn));
        }
        __syncthreads();
    }

    // ---- Output head: out = h_last @ W_out^T + b_out (16x10, tiny) -----------
    if (tid < BM * LSTM_O) {
        const int m = tid / LSTM_O;
        const int o = tid % LSTM_O;
        float s = b_out[o];
        #pragma unroll 8
        for (int k = 0; k < LSTM_H; ++k)
            s += (float)h16[m][k] * W_out[o * LSTM_H + k];
        out[(size_t)(b0 + m) * LSTM_O + o] = s;
    }
}

extern "C" void kernel_launch(void* const* d_in, const int* in_sizes, int n_in,
                              void* d_out, int out_size, void* d_ws, size_t ws_size,
                              hipStream_t stream) {
    const float* x     = (const float*)d_in[0];
    const float* W_ih  = (const float*)d_in[1];
    const float* W_hh  = (const float*)d_in[2];
    const float* b_ih  = (const float*)d_in[3];
    const float* b_hh  = (const float*)d_in[4];
    const float* W_out = (const float*)d_in[5];
    const float* b_out = (const float*)d_in[6];
    float* out = (float*)d_out;

    dim3 grid(LSTM_B / BM);   // 2048 blocks, 16 batch rows each
    dim3 block(512);          // 16 waves (wave32): one gate-column tile per wave
    lstm_wmma_kernel<<<grid, block, 0, stream>>>(x, W_ih, W_hh, b_ih, b_hh,
                                                 W_out, b_out, out);
}
